// ViSNet_9363028705442
// MI455X (gfx1250) — matrix operand order repaired and checked by reading, not compiled, not run on hardware
//
#include <hip/hip_runtime.h>
#include <math.h>

// ViSNet forward for MI455X (gfx1250, wave32, WMMA).
// All dense GEMMs use v_wmma_f32_16x16x32_f16 with operands pre-packed into
// the CDNA5 ISA 7.12.2 fragment layouts (f16), so the GEMM inner loop is
// pure 128-bit loads + WMMA.

#define NNODES  4096
#define NEDGES  131072
#define HIDD    128
#define NRBF    32
#define NLAYERS 4
#define NGRAPH  32
#define VNODEZ  119
#define CUTR    5.0f
#define PI_F    3.14159265358979323846f

typedef __attribute__((ext_vector_type(16))) _Float16 v16h;
typedef __attribute__((ext_vector_type(8)))  float    v8f;

__device__ __forceinline__ float dsilu(float x) { return x / (1.0f + __expf(-x)); }
__device__ __forceinline__ float dcut(float r) {
  return (r < CUTR) ? 0.5f * (__cosf(r * (PI_F / CUTR)) + 1.0f) : 0.0f;
}

enum { EPI_STORE = 0, EPI_SILU = 1, EPI_NB = 2, EPI_EDGEATTR = 3, EPI_SILUMUL = 4 };

// ---------------------------------------------------------------------------
// Operand packing into WMMA fragment layouts (f16).
// A fragment (16x32, per k-tile): lane L = (row%16) + 16*(khalf), element h ->
//   K = ktile*32 + (h<8?0:16) + 8*(L>>4) + (h&7).
// B fragment (32x16, per n-tile): lane L = (col%16) + 16*(khalf), element h ->
//   K = ktile*32 + (h<8?0:16) + 8*(L>>4) + (h&7).
// Packed storage: fragment-contiguous, 32 lanes x 16 halves.
// ---------------------------------------------------------------------------
__global__ __launch_bounds__(32) void k_packA(const float* __restrict__ A0,
                                              const float* __restrict__ A1,
                                              int K, int ksplit,
                                              _Float16* __restrict__ pA)
{
  const int lane = threadIdx.x;
  const int mt = blockIdx.x, kt = blockIdx.y;
  const int KT = gridDim.y;
  const int row = mt * 16 + (lane & 15);
  const int kh  = (lane >> 4) * 8;
  const int k0  = kt * 32;
  // ksplit is always a multiple of 32 here, so a k-tile never spans A0/A1.
  const float* S; int lda, kb;
  if (k0 < ksplit) { S = A0; lda = ksplit;     kb = k0; }
  else             { S = A1; lda = K - ksplit; kb = k0 - ksplit; }
  const float* r1 = S + (size_t)row * lda + kb + kh;       // K run [k0+kh, +8)
  const float* r2 = r1 + 16;                               // K run [k0+16+kh, +8)
  v16h a;
#pragma unroll
  for (int h = 0; h < 8; ++h) a[h] = (_Float16)r1[h];
#pragma unroll
  for (int h = 0; h < 8; ++h) a[8 + h] = (_Float16)r2[h];
  *(v16h*)(pA + ((size_t)(mt * KT + kt) * 32 + lane) * 16) = a;
}

__global__ __launch_bounds__(32) void k_packW(const float* __restrict__ W, int NOUT,
                                              _Float16* __restrict__ pW)
{
  const int lane = threadIdx.x;
  const int kt = blockIdx.x, nt = blockIdx.y;
  const int NT = gridDim.y;
  const int n  = nt * 16 + (lane & 15);
  const int kb = kt * 32 + (lane >> 4) * 8;
  v16h b;
#pragma unroll
  for (int h = 0; h < 16; ++h) b[h] = (_Float16)W[(size_t)(kb + h + ((h >> 3) << 3)) * NOUT + n];
  *(v16h*)(pW + ((size_t)(kt * NT + nt) * 32 + lane) * 16) = b;
}

// ---------------------------------------------------------------------------
// WMMA GEMM: Y(M,NOUT) = epi( A @ W + bias ), operands pre-packed f16.
// One wave per block; 16(M) x 64(N) tile; 4 WMMAs per k-step, A reused 4x.
// ---------------------------------------------------------------------------
template<int EPI>
__global__ __launch_bounds__(32) void wmma_gemm_k(
    const _Float16* __restrict__ pA, const _Float16* __restrict__ pB,
    int KT, int NT, const float* __restrict__ bias,
    float* __restrict__ Y, int NOUT,
    const float* __restrict__ auxR,   // EPI_NB: per-row scale (C * nonself)
    const float* __restrict__ auxM,   // EPI_EDGEATTR: x; EPI_SILUMUL: w_dot
    const float* __restrict__ tab,    // EPI_NB: nb_emb[z[node]] per node
    const int* __restrict__ sidx, const int* __restrict__ didx)
{
  const int lane = threadIdx.x;
  const int mt   = blockIdx.x;
  const int nt0  = blockIdx.y * 4;
  v8f acc[4] = {{}, {}, {}, {}};

  const _Float16* pa = pA + ((size_t)mt * KT) * 512 + lane * 16;
  for (int kt = 0; kt < KT; ++kt) {
    const v16h a = *(const v16h*)pa;
    pa += 512;
    const _Float16* pb = pB + ((size_t)kt * NT + nt0) * 512 + lane * 16;
    const v16h b0 = *(const v16h*)(pb);
    const v16h b1 = *(const v16h*)(pb + 512);
    const v16h b2 = *(const v16h*)(pb + 1024);
    const v16h b3 = *(const v16h*)(pb + 1536);
    acc[0] = __builtin_amdgcn_wmma_f32_16x16x32_f16(false, a, false, b0, (short)0, acc[0], false, false);
    acc[1] = __builtin_amdgcn_wmma_f32_16x16x32_f16(false, a, false, b1, (short)0, acc[1], false, false);
    acc[2] = __builtin_amdgcn_wmma_f32_16x16x32_f16(false, a, false, b2, (short)0, acc[2], false, false);
    acc[3] = __builtin_amdgcn_wmma_f32_16x16x32_f16(false, a, false, b3, (short)0, acc[3], false, false);
  }

  // C/D layout: VGPR r -> row mt*16 + r + 8*(lane>>4), col (nt0+t)*16 + (lane&15).
  const int mlane = lane & 15;
  const int hi    = lane >> 4;
#pragma unroll
  for (int t = 0; t < 4; ++t) {
    const int n = (nt0 + t) * 16 + mlane;
    const float bv = bias ? bias[n] : 0.0f;
#pragma unroll
    for (int r = 0; r < 8; ++r) {
      const int row = mt * 16 + r + hi * 8;
      const float val = acc[t][r] + bv;
      if (EPI == EPI_STORE) {
        Y[(size_t)row * NOUT + n] = val;
      } else if (EPI == EPI_SILU) {
        Y[(size_t)row * NOUT + n] = dsilu(val);
      } else if (EPI == EPI_NB) {
        const float wv = val * auxR[row];
        atomicAdd(&Y[(size_t)didx[row] * HIDD + n], tab[(size_t)sidx[row] * HIDD + n] * wv);
      } else if (EPI == EPI_EDGEATTR) {
        Y[(size_t)row * NOUT + n] =
            val * (auxM[(size_t)sidx[row] * HIDD + n] + auxM[(size_t)didx[row] * HIDD + n]);
      } else if (EPI == EPI_SILUMUL) {
        Y[(size_t)row * NOUT + n] = dsilu(val) * auxM[(size_t)row * NOUT + n];
      }
    }
  }
}

// ---------------------------------------------------------------------------
// Elementwise / reduction kernels
// ---------------------------------------------------------------------------
// Geometry; writes RBF directly in packed A-fragment layout (K=32 -> KT=1).
__global__ void k_geom(const float* __restrict__ pos,
                       const int* __restrict__ src, const int* __restrict__ dst,
                       float* __restrict__ cutv, float* __restrict__ cn,
                       float* __restrict__ dij, _Float16* __restrict__ pRbf)
{
  const int e = blockIdx.x * blockDim.x + threadIdx.x;
  if (e >= NEDGES) return;
  const int s = src[e], d = dst[e];
  const float vx = pos[s * 3 + 0] - pos[d * 3 + 0];
  const float vy = pos[s * 3 + 1] - pos[d * 3 + 1];
  const float vz = pos[s * 3 + 2] - pos[d * 3 + 2];
  const float r = sqrtf(fmaxf(vx * vx + vy * vy + vz * vz, 1e-12f));
  const float C = dcut(r);
  const bool nonself = (s != d);
  cutv[e] = C;
  cn[e]   = nonself ? C : 0.0f;
  const float inv = nonself ? (1.0f / r) : 1.0f;
  dij[e * 3 + 0] = vx * inv;
  dij[e * 3 + 1] = vy * inv;
  dij[e * 3 + 2] = vz * inv;
  const float e5 = 0.006737946999085467f;   // exp(-5)
  const float w  = 0.0625f * (1.0f - e5);
  const float beta = 1.0f / (w * w);
  const float ed = __expf(-r);
  const size_t mtbase = (size_t)(e >> 4) * 32;
#pragma unroll
  for (int i = 0; i < NRBF; ++i) {
    const float m = e5 + (float)i * (1.0f - e5) / 31.0f;
    const float t = ed - m;
    const float rv = C * __expf(-beta * t * t);
    const int L = (e & 15) + (((i & 15) >= 8) ? 16 : 0);
    const int h = ((i >= 16) ? 8 : 0) + (i & 7);
    pRbf[(mtbase + L) * 16 + h] = (_Float16)rv;
  }
}

__global__ void k_gather_emb(const float* __restrict__ emb, const float* __restrict__ nb_emb,
                             const int* __restrict__ z,
                             float* __restrict__ x0, float* __restrict__ nbz)
{
  const int i = blockIdx.x * blockDim.x + threadIdx.x;
  if (i >= NNODES * HIDD) return;
  const int n = i >> 7, c = i & 127;
  const int zz = z[n];
  x0[i]  = emb[(size_t)zz * HIDD + c];
  nbz[i] = nb_emb[(size_t)zz * HIDD + c];
}

// LayerNorm over 128 features: one wave per row.
__global__ __launch_bounds__(32) void k_ln(const float* __restrict__ X,
                                           const float* __restrict__ g, const float* __restrict__ b,
                                           float* __restrict__ Y, int rows)
{
  const int row = blockIdx.x;
  if (row >= rows) return;
  const int lane = threadIdx.x;
  float v[4];
  float s = 0.0f;
#pragma unroll
  for (int j = 0; j < 4; ++j) { v[j] = X[(size_t)row * HIDD + lane + 32 * j]; s += v[j]; }
#pragma unroll
  for (int off = 16; off > 0; off >>= 1) s += __shfl_xor(s, off);
  const float mean = s * (1.0f / HIDD);
  float q = 0.0f;
#pragma unroll
  for (int j = 0; j < 4; ++j) { const float t = v[j] - mean; q += t * t; }
#pragma unroll
  for (int off = 16; off > 0; off >>= 1) q += __shfl_xor(q, off);
  const float inv = rsqrtf(q * (1.0f / HIDD) + 1e-5f);
#pragma unroll
  for (int j = 0; j < 4; ++j) {
    const int c = lane + 32 * j;
    Y[(size_t)row * HIDD + c] = (v[j] - mean) * inv * g[c] + b[c];
  }
}

__global__ void k_vln(const float* __restrict__ vec, const float* __restrict__ w,
                      float* __restrict__ vecs)
{
  const int i = blockIdx.x * blockDim.x + threadIdx.x;
  if (i >= NNODES * 3 * HIDD) return;
  vecs[i] = vec[i] * w[i & 127];
}

__global__ void k_vecdot(const float* __restrict__ vecW, float* __restrict__ vdot)
{
  const int i = blockIdx.x * blockDim.x + threadIdx.x;
  if (i >= NNODES * HIDD) return;
  const int n = i >> 7, c = i & 127;
  float s = 0.0f;
#pragma unroll
  for (int comp = 0; comp < 3; ++comp) {
    const size_t r = (size_t)(n * 3 + comp) * 384;
    s += vecW[r + c] * vecW[r + 128 + c];
  }
  vdot[i] = s;
}

// Attention + value message. Writes the message directly in packed
// A-fragment layout (pMsg) for the Ws GEMM, and scatters x_agg.
__global__ __launch_bounds__(128) void k_attn(
    const float* __restrict__ q, const float* __restrict__ k, const float* __restrict__ v,
    const float* __restrict__ dk, const float* __restrict__ dv,
    const float* __restrict__ cutv,
    const int* __restrict__ src, const int* __restrict__ dst,
    _Float16* __restrict__ pMsg, float* __restrict__ xagg)
{
  const int e = blockIdx.x;
  const int c = threadIdx.x;
  const int s = src[e], d = dst[e];
  float t = q[(size_t)d * HIDD + c] * k[(size_t)s * HIDD + c] * dk[(size_t)e * HIDD + c];
#pragma unroll
  for (int off = 8; off > 0; off >>= 1) t += __shfl_xor(t, off);   // per-head sum
  const float attn = dsilu(t) * cutv[e];
  const float m = v[(size_t)s * HIDD + c] * dv[(size_t)e * HIDD + c] * attn;
  const int kt = c >> 5, kr = c & 31;
  const int L = (e & 15) + (((kr & 15) >= 8) ? 16 : 0);
  const int h = ((kr >= 16) ? 8 : 0) + (kr & 7);
  pMsg[(((size_t)(e >> 4) * 4 + kt) * 32 + L) * 16 + h] = (_Float16)m;
  atomicAdd(&xagg[(size_t)d * HIDD + c], m);
}

__global__ void k_vecmsg(const float* __restrict__ vecs, const float* __restrict__ sY,
                         const float* __restrict__ dij,
                         const int* __restrict__ src, const int* __restrict__ dst,
                         float* __restrict__ vecagg)
{
  const long long i = (long long)blockIdx.x * blockDim.x + threadIdx.x;
  if (i >= (long long)NEDGES * 384) return;
  const int e = (int)(i / 384);
  const int rem = (int)(i % 384);
  const int comp = rem >> 7, c = rem & 127;
  const float s1 = sY[(size_t)e * 256 + c];
  const float s2 = sY[(size_t)e * 256 + 128 + c];
  const float vm = vecs[(size_t)(src[e] * 3 + comp) * HIDD + c] * s1 + s2 * dij[e * 3 + comp];
  atomicAdd(&vecagg[(size_t)(dst[e] * 3 + comp) * HIDD + c], vm);
}

__global__ void k_update(float* __restrict__ x, float* __restrict__ vec,
                         const float* __restrict__ vdot, const float* __restrict__ oY,
                         const float* __restrict__ vecW, const float* __restrict__ vecagg)
{
  const int i = blockIdx.x * blockDim.x + threadIdx.x;
  if (i >= NNODES * HIDD) return;
  const int n = i >> 7, c = i & 127;
  const float o1 = oY[(size_t)n * 384 + c];
  const float o2 = oY[(size_t)n * 384 + 128 + c];
  const float o3 = oY[(size_t)n * 384 + 256 + c];
  x[i] += vdot[i] * o2 + o3;
#pragma unroll
  for (int comp = 0; comp < 3; ++comp) {
    const size_t vi = (size_t)(n * 3 + comp) * HIDD + c;
    vec[vi] += vecW[(size_t)(n * 3 + comp) * 384 + 256 + c] * o1 + vecagg[vi];
  }
}

__global__ void k_wdot(const float* __restrict__ t1, const float* __restrict__ t2,
                       const float* __restrict__ dij,
                       const int* __restrict__ src, const int* __restrict__ dst,
                       float* __restrict__ wdot)
{
  const long long i = (long long)blockIdx.x * blockDim.x + threadIdx.x;
  if (i >= (long long)NEDGES * HIDD) return;
  const int e = (int)(i >> 7), c = (int)(i & 127);
  const int s = src[e], d = dst[e];
  const float dx = dij[e * 3 + 0], dy = dij[e * 3 + 1], dz = dij[e * 3 + 2];
  const float ax = t1[(size_t)(d * 3 + 0) * HIDD + c];
  const float ay = t1[(size_t)(d * 3 + 1) * HIDD + c];
  const float az = t1[(size_t)(d * 3 + 2) * HIDD + c];
  const float p1 = ax * dx + ay * dy + az * dz;
  const float w1x = ax - p1 * dx, w1y = ay - p1 * dy, w1z = az - p1 * dz;
  const float bx = t2[(size_t)(s * 3 + 0) * HIDD + c];
  const float by = t2[(size_t)(s * 3 + 1) * HIDD + c];
  const float bz = t2[(size_t)(s * 3 + 2) * HIDD + c];
  const float p2 = bx * dx + by * dy + bz * dz;   // reject(v,-d) == v - (v.d)d
  const float w2x = bx - p2 * dx, w2y = by - p2 * dy, w2z = bz - p2 * dz;
  wdot[i] = w1x * w2x + w1y * w2y + w1z * w2z;
}

__global__ void k_add(float* __restrict__ Y, const float* __restrict__ Xv, long long n)
{
  const long long i = (long long)blockIdx.x * blockDim.x + threadIdx.x;
  if (i < n) Y[i] += Xv[i];
}

__global__ void k_pool(const float* __restrict__ xout, const int* __restrict__ z,
                       const int* __restrict__ batch, float* __restrict__ out)
{
  const int i = blockIdx.x * blockDim.x + threadIdx.x;
  if (i >= NNODES * HIDD) return;
  const int n = i >> 7, c = i & 127;
  const size_t base = (size_t)NNODES * HIDD;
  const size_t off = (z[n] == VNODEZ) ? base : base + (size_t)NGRAPH * HIDD;
  atomicAdd(&out[off + (size_t)batch[n] * HIDD + c], xout[i]);
}

// ---------------------------------------------------------------------------
// Host side
// ---------------------------------------------------------------------------
static void packA(hipStream_t st, const float* A0, const float* A1, int K, int ksplit,
                  int M, _Float16* pA)
{
  hipLaunchKernelGGL(k_packA, dim3(M / 16, K / 32), dim3(32), 0, st, A0, A1, K, ksplit, pA);
}

static void gemm(hipStream_t st, int epi, const _Float16* pA,
                 const float* W, _Float16* pW, int K, int NOUT,
                 const float* bias, float* Y, int M,
                 const float* auxR, const float* auxM, const float* tab,
                 const int* sidx, const int* didx)
{
  hipLaunchKernelGGL(k_packW, dim3(K / 32, NOUT / 16), dim3(32), 0, st, W, NOUT, pW);
  const int KT = K / 32, NT = NOUT / 16;
  dim3 g(M / 16, NOUT / 64), b(32);
  switch (epi) {
    case EPI_STORE:    wmma_gemm_k<EPI_STORE>   <<<g, b, 0, st>>>(pA, pW, KT, NT, bias, Y, NOUT, auxR, auxM, tab, sidx, didx); break;
    case EPI_SILU:     wmma_gemm_k<EPI_SILU>    <<<g, b, 0, st>>>(pA, pW, KT, NT, bias, Y, NOUT, auxR, auxM, tab, sidx, didx); break;
    case EPI_NB:       wmma_gemm_k<EPI_NB>      <<<g, b, 0, st>>>(pA, pW, KT, NT, bias, Y, NOUT, auxR, auxM, tab, sidx, didx); break;
    case EPI_EDGEATTR: wmma_gemm_k<EPI_EDGEATTR><<<g, b, 0, st>>>(pA, pW, KT, NT, bias, Y, NOUT, auxR, auxM, tab, sidx, didx); break;
    default:           wmma_gemm_k<EPI_SILUMUL> <<<g, b, 0, st>>>(pA, pW, KT, NT, bias, Y, NOUT, auxR, auxM, tab, sidx, didx); break;
  }
}

extern "C" void kernel_launch(void* const* d_in, const int* in_sizes, int n_in,
                              void* d_out, int out_size, void* d_ws, size_t ws_size,
                              hipStream_t stream)
{
  (void)in_sizes; (void)n_in; (void)out_size; (void)ws_size;
  // ---- input mapping (pytree: pos, z, batch, edge_index, params{sorted keys})
  const float* pos  = (const float*)d_in[0];
  const int*   z    = (const int*)d_in[1];
  const int*   batch= (const int*)d_in[2];
  const int*   src  = (const int*)d_in[3];
  const int*   dst  = ((const int*)d_in[3]) + NEDGES;
  const float* edge_W    = (const float*)d_in[4];
  const float* edge_b    = (const float*)d_in[5];
  const float* emb       = (const float*)d_in[6];
  const float* nb_comb_W = (const float*)d_in[7 + 22 * NLAYERS + 0];
  const float* nb_comb_b = (const float*)d_in[7 + 22 * NLAYERS + 1];
  const float* nb_dist_W = (const float*)d_in[7 + 22 * NLAYERS + 2];
  const float* nb_dist_b = (const float*)d_in[7 + 22 * NLAYERS + 3];
  const float* nb_emb    = (const float*)d_in[7 + 22 * NLAYERS + 4];
  const float* out_ln_b  = (const float*)d_in[7 + 22 * NLAYERS + 5];
  const float* out_ln_g  = (const float*)d_in[7 + 22 * NLAYERS + 6];

  // ---- workspace arena (floats; all sizes multiples of 8 -> 32B aligned)
  float* w = (float*)d_ws;
  size_t off = 0;
  auto alloc = [&](size_t nfl) { float* p = w + off; off += (nfl + 7) & ~(size_t)7; return p; };
  const size_t NH = (size_t)NNODES * HIDD;
  const size_t VH = (size_t)NNODES * 3 * HIDD;
  const size_t EH = (size_t)NEDGES * HIDD;

  float* cutv  = alloc(NEDGES);
  float* cn    = alloc(NEDGES);
  float* dij   = alloc((size_t)NEDGES * 3);
  float* ea    = alloc(EH);             // edge_attr (f32, persistent)
  float* bufAB = alloc(2 * EH);         // dk|dv -> s -> df
  float* bufC  = alloc(EH);             // w_dot
  float* x0    = alloc(NH);
  float* nbz   = alloc(NH);
  float* xnb   = alloc(NH);
  float* x     = alloc(NH);
  float* xln   = alloc(NH);
  float* vdot  = alloc(NH);
  float* q     = alloc(NH);
  float* kk    = alloc(NH);
  float* vv    = alloc(NH);
  float* xagg  = alloc(NH);
  float* vec   = alloc(VH);
  float* vecs  = alloc(VH);
  float* vecagg= alloc(VH);
  float* t1    = alloc(VH);
  float* t2    = alloc(VH);
  float* oY    = alloc((size_t)NNODES * 384);
  float* vecW  = alloc((size_t)NNODES * 3 * 384);
  // packed f16 operand buffers (sized in halves -> /2 floats)
  _Float16* pRbf  = (_Float16*)alloc((size_t)NEDGES * NRBF / 2);
  _Float16* pEA   = (_Float16*)alloc(EH / 2);
  _Float16* pMsg  = (_Float16*)alloc(EH / 2);
  _Float16* pXln  = (_Float16*)alloc(NH / 2);
  _Float16* pVecs = (_Float16*)alloc(VH / 2);
  _Float16* pXagg = (_Float16*)alloc(NH / 2);
  _Float16* pCat  = (_Float16*)alloc((size_t)NNODES * 256 / 2);
  _Float16* pW    = (_Float16*)alloc((size_t)HIDD * 384 / 2);

  float* out = (float*)d_out;
  const int TB = 256;

  // ---- stage 0: geometry + embeddings
  hipLaunchKernelGGL(k_geom, dim3((NEDGES + TB - 1) / TB), dim3(TB), 0, stream,
                     pos, src, dst, cutv, cn, dij, pRbf);
  hipLaunchKernelGGL(k_gather_emb, dim3((int)(NH / TB)), dim3(TB), 0, stream,
                     emb, nb_emb, z, x0, nbz);
  hipMemsetAsync(xnb, 0, NH * sizeof(float), stream);
  hipMemsetAsync(vec, 0, VH * sizeof(float), stream);
  // x_nb[dst] += nb_emb[z[src]] * ((rbf@Wd+b)*C*nonself)
  gemm(stream, EPI_NB, pRbf, nb_dist_W, pW, NRBF, HIDD, nb_dist_b,
       xnb, NEDGES, cn, nullptr, nbz, src, dst);
  // x = concat(x0, x_nb) @ nb_comb_W + b
  packA(stream, x0, xnb, 2 * HIDD, HIDD, NNODES, pCat);
  gemm(stream, EPI_STORE, pCat, nb_comb_W, pW, 2 * HIDD, HIDD, nb_comb_b,
       x, NNODES, 0, 0, 0, 0, 0);
  // edge_attr = (x[src]+x[dst]) * (rbf@edge_W + edge_b)
  gemm(stream, EPI_EDGEATTR, pRbf, edge_W, pW, NRBF, HIDD, edge_b,
       ea, NEDGES, nullptr, x, nullptr, src, dst);

  // ---- layers
  for (int li = 0; li < NLAYERS; ++li) {
    const int B = 7 + 22 * li;
    const float* Wdk = (const float*)d_in[B + 0];
    const float* Wdv = (const float*)d_in[B + 1];
    const float* Wf  = (const float*)d_in[B + 2];
    const float* Wk  = (const float*)d_in[B + 3];
    const float* Wo  = (const float*)d_in[B + 4];
    const float* Wq  = (const float*)d_in[B + 5];
    const float* Ws  = (const float*)d_in[B + 6];
    const float* Wv  = (const float*)d_in[B + 7];
    const float* Wvec= (const float*)d_in[B + 8];
    const float* Wws = (const float*)d_in[B + 9];
    const float* Wwt = (const float*)d_in[B + 10];
    const float* bdk = (const float*)d_in[B + 11];
    const float* bdv = (const float*)d_in[B + 12];
    const float* bf  = (const float*)d_in[B + 13];
    const float* bk  = (const float*)d_in[B + 14];
    const float* bo  = (const float*)d_in[B + 15];
    const float* bq  = (const float*)d_in[B + 16];
    const float* bs  = (const float*)d_in[B + 17];
    const float* bv  = (const float*)d_in[B + 18];
    const float* lnb = (const float*)d_in[B + 19];
    const float* lng = (const float*)d_in[B + 20];
    const float* vlnw= (const float*)d_in[B + 21];
    const bool last = (li == NLAYERS - 1);

    hipLaunchKernelGGL(k_ln, dim3(NNODES), dim3(32), 0, stream, x, lng, lnb, xln, NNODES);
    hipLaunchKernelGGL(k_vln, dim3((int)(VH / TB)), dim3(TB), 0, stream, vec, vlnw, vecs);
    packA(stream, xln, nullptr, HIDD, HIDD, NNODES, pXln);
    packA(stream, vecs, nullptr, HIDD, HIDD, NNODES * 3, pVecs);
    packA(stream, ea, nullptr, HIDD, HIDD, NEDGES, pEA);

    gemm(stream, EPI_STORE, pXln, Wq, pW, HIDD, HIDD, bq, q,  NNODES, 0, 0, 0, 0, 0);
    gemm(stream, EPI_STORE, pXln, Wk, pW, HIDD, HIDD, bk, kk, NNODES, 0, 0, 0, 0, 0);
    gemm(stream, EPI_STORE, pXln, Wv, pW, HIDD, HIDD, bv, vv, NNODES, 0, 0, 0, 0, 0);
    float* dk = bufAB;
    float* dv = bufAB + EH;
    gemm(stream, EPI_SILU, pEA, Wdk, pW, HIDD, HIDD, bdk, dk, NEDGES, 0, 0, 0, 0, 0);
    gemm(stream, EPI_SILU, pEA, Wdv, pW, HIDD, HIDD, bdv, dv, NEDGES, 0, 0, 0, 0, 0);
    gemm(stream, EPI_STORE, pVecs, Wvec, pW, HIDD, 384, nullptr, vecW, NNODES * 3, 0, 0, 0, 0, 0);
    hipLaunchKernelGGL(k_vecdot, dim3((int)(NH / TB)), dim3(TB), 0, stream, vecW, vdot);

    hipMemsetAsync(xagg, 0, NH * sizeof(float), stream);
    hipLaunchKernelGGL(k_attn, dim3(NEDGES), dim3(HIDD), 0, stream,
                       q, kk, vv, dk, dv, cutv, src, dst, pMsg, xagg);
    float* sY = bufAB;                         // dk/dv dead; s is [E,256]
    gemm(stream, EPI_SILU, pMsg, Ws, pW, HIDD, 256, bs, sY, NEDGES, 0, 0, 0, 0, 0);
    hipMemsetAsync(vecagg, 0, VH * sizeof(float), stream);
    {
      long long tot = (long long)NEDGES * 384;
      hipLaunchKernelGGL(k_vecmsg, dim3((int)((tot + TB - 1) / TB)), dim3(TB), 0, stream,
                         vecs, sY, dij, src, dst, vecagg);
    }
    packA(stream, xagg, nullptr, HIDD, HIDD, NNODES, pXagg);
    gemm(stream, EPI_STORE, pXagg, Wo, pW, HIDD, 384, bo, oY, NNODES, 0, 0, 0, 0, 0);
    hipLaunchKernelGGL(k_update, dim3((int)(NH / TB)), dim3(TB), 0, stream,
                       x, vec, vdot, oY, vecW, vecagg);

    if (!last) {
      gemm(stream, EPI_STORE, pVecs, Wwt, pW, HIDD, HIDD, nullptr, t1, NNODES * 3, 0, 0, 0, 0, 0);
      gemm(stream, EPI_STORE, pVecs, Wws, pW, HIDD, HIDD, nullptr, t2, NNODES * 3, 0, 0, 0, 0, 0);
      float* wdot = bufC;
      hipLaunchKernelGGL(k_wdot, dim3((int)(EH / TB)), dim3(TB), 0, stream,
                         t1, t2, dij, src, dst, wdot);
      float* df = bufAB;                       // s dead
      gemm(stream, EPI_SILUMUL, pEA, Wf, pW, HIDD, HIDD, bf, df, NEDGES,
           nullptr, wdot, nullptr, nullptr, nullptr);
      hipLaunchKernelGGL(k_add, dim3((int)(EH / TB)), dim3(TB), 0, stream,
                         ea, df, (long long)EH);
    }
  }

  // ---- output: out-LN into d_out, then pooled sums
  hipLaunchKernelGGL(k_ln, dim3(NNODES), dim3(32), 0, stream, x, out_ln_g, out_ln_b, out, NNODES);
  hipMemsetAsync(out + NH, 0, (size_t)2 * NGRAPH * HIDD * sizeof(float), stream);
  hipLaunchKernelGGL(k_pool, dim3((int)(NH / TB)), dim3(TB), 0, stream, out, z, batch, out);
}
